// GPTSelfAttention_14800457302465
// MI455X (gfx1250) — compile-verified, matrix operand order checked by reading, not run on hardware
//
#include <hip/hip_runtime.h>
#include <hip/hip_bf16.h>

// ---------------------------------------------------------------------------
// GPT self-attention forward for MI455X (gfx1250, wave32, WMMA).
//   out = (ctx [B,S,HID] fp32, probs [B,NH,S,S] fp32), concatenated flat.
// The workload is bound by the 537MB probs write stream (23.3 TB/s HBM ->
// ~23us floor); compute (~60 GFLOP) is tiny for the WMMA pipes. So: f16 WMMA
// everywhere, probs written exactly once as full-width b128 stores, scores
// never round-trip HBM (two-pass online softmax), K/V fragments software-
// pipelined so v_wmma doesn't sit behind s_wait_loadcnt 0 drains.
// ---------------------------------------------------------------------------

#define DEV __device__ __forceinline__

typedef __attribute__((ext_vector_type(16))) _Float16 v16h;
typedef __attribute__((ext_vector_type(8)))  float    v8f;

union FragH { v16h v; uint4 q[2]; };

static constexpr int Bc = 2, Sc = 2048, HIDc = 1024, NHc = 16, HDc = 64;
static constexpr int STF = 36;   // f32 staging row stride (floats, 16B-aligned)

// A/B fragment for V_WMMA_F32_16X16X32_F16, operand stored row-major along K
// (A: row m = lane&15; B: "row" n = lane&15 of the K-major matrix).
// Per-lane K offsets: base = (lane>=16 ? 8 : 0); halves {0..7, 16..23}.
DEV v16h load_frag_g(const _Float16* __restrict__ base, int row0, int ld,
                     int k0, int lane) {
  const _Float16* p = base + (size_t)(row0 + (lane & 15)) * ld + k0 + ((lane >> 4) << 3);
  FragH f;
  f.q[0] = *(const uint4*)(p);        // K = base+0..7
  f.q[1] = *(const uint4*)(p + 16);   // K = base+16..23
  return f.v;
}

// Build the same A-fragment pattern from a per-wave f32 LDS staging tile
// (16 rows x 32 cols, row stride STF floats), converting f32 -> f16.
DEV v16h load_frag_lds_f32(const float* p0, int lane) {
  const float* p = p0 + (lane & 15) * STF + ((lane >> 4) << 3);
  float4 x0 = *(const float4*)(p);        // K = base+0..3
  float4 x1 = *(const float4*)(p + 4);    // K = base+4..7
  float4 x2 = *(const float4*)(p + 16);   // K = base+16..19
  float4 x3 = *(const float4*)(p + 20);   // K = base+20..23
  v16h v;
  v[0]  = (_Float16)x0.x; v[1]  = (_Float16)x0.y;
  v[2]  = (_Float16)x0.z; v[3]  = (_Float16)x0.w;
  v[4]  = (_Float16)x1.x; v[5]  = (_Float16)x1.y;
  v[6]  = (_Float16)x1.z; v[7]  = (_Float16)x1.w;
  v[8]  = (_Float16)x2.x; v[9]  = (_Float16)x2.y;
  v[10] = (_Float16)x2.z; v[11] = (_Float16)x2.w;
  v[12] = (_Float16)x3.x; v[13] = (_Float16)x3.y;
  v[14] = (_Float16)x3.z; v[15] = (_Float16)x3.w;
  return v;
}

__global__ void cvt_f32_to_f16(const float* __restrict__ src,
                               _Float16* __restrict__ dst, int n) {
  int i = blockIdx.x * blockDim.x + threadIdx.x;
  if (i < n) dst[i] = (_Float16)src[i];
}

// ---------------------------------------------------------------------------
// QKV projection: one wave computes a 16x64 tile of X @ W^T + b.
// Q,K stored [B,NH,S,HD] f16; V stored transposed [B,NH,HD,S] f16.
// ---------------------------------------------------------------------------
__global__ __launch_bounds__(128) void qkv_proj(
    const _Float16* __restrict__ Xh,
    const _Float16* __restrict__ Wqh, const _Float16* __restrict__ Wkh,
    const _Float16* __restrict__ Wvh,
    const float* __restrict__ bq, const float* __restrict__ bk,
    const float* __restrict__ bv,
    _Float16* __restrict__ Qh, _Float16* __restrict__ Kh,
    _Float16* __restrict__ Vth) {
  int lane = threadIdx.x & 31;
  int wid  = blockIdx.x * (blockDim.x >> 5) + (threadIdx.x >> 5);  // 0..12287
  int mat  = wid / 4096;            // 0=Q 1=K 2=V
  int rem  = wid - mat * 4096;
  int m0   = (rem >> 4) * 16;       // row tile in [0,4096)
  int h    = rem & 15;              // head == 64-wide N tile

  const _Float16* W    = (mat == 0) ? Wqh : (mat == 1) ? Wkh : Wvh;
  const float*    bias = (mat == 0) ? bq  : (mat == 1) ? bk  : bv;

  v8f acc[4] = {};
  v16h a = load_frag_g(Xh, m0, HIDc, 0, lane);     // pipelined A fragment
  for (int k0 = 0; k0 < HIDc; k0 += 32) {
    int kn = (k0 + 32) & (HIDc - 1);
    v16h an = load_frag_g(Xh, m0, HIDc, kn, lane); // prefetch next A
#pragma unroll
    for (int t = 0; t < 4; ++t) {
      v16h b = load_frag_g(W, h * 64 + t * 16, HIDc, k0, lane);
      acc[t] = __builtin_amdgcn_wmma_f32_16x16x32_f16(
          false, a, false, b, (short)0, acc[t], false, false);
    }
    a = an;
  }

  int bb  = m0 >> 11;               // batch
  int s0  = m0 & 2047;              // sequence base
  int hi8 = (lane >> 4) << 3;       // C layout: lanes>=16 hold rows m+8
  int nl  = lane & 15;
#pragma unroll
  for (int t = 0; t < 4; ++t) {
    float bconst = bias[h * 64 + t * 16 + nl];
#pragma unroll
    for (int r = 0; r < 8; ++r) {
      float val = acc[t][r] + bconst;
      int s = s0 + r + hi8;
      int d = t * 16 + nl;
      if (mat == 0)
        Qh[(size_t)((bb * NHc + h) * Sc + s) * HDc + d] = (_Float16)val;
      else if (mat == 1)
        Kh[(size_t)((bb * NHc + h) * Sc + s) * HDc + d] = (_Float16)val;
      else
        Vth[(size_t)((bb * NHc + h) * HDc + d) * Sc + s] = (_Float16)val;
    }
  }
}

// ---------------------------------------------------------------------------
// Attention: one wave owns (b, h, 16-row q tile); no inter-wave sync.
// Pass 1: online softmax stats over recomputed WMMA score tiles (K frags
//         double-buffered so wmma overlaps the next tile's loads).
// Pass 2: recompute scores, stage normalized f32 probs in LDS, drain them as
//         row-contiguous global_store_b128 (the HBM-bound stream), rebuild
//         the f16 A-fragment from the same LDS tile, accumulate ctx = P @ V.
// ---------------------------------------------------------------------------
__global__ __launch_bounds__(256) void attn_fused(
    const _Float16* __restrict__ Qh, const _Float16* __restrict__ Kh,
    const _Float16* __restrict__ Vth, const float* __restrict__ amask,
    float* __restrict__ ctx, float* __restrict__ probs) {
  __shared__ float stage[8][16 * STF];     // per-wave 16x32 f32 tile (padded)
  int lane = threadIdx.x & 31;
  int w    = threadIdx.x >> 5;
  int wid  = blockIdx.x * 8 + w;           // 0..4095
  int b    = wid >> 11;
  int rem  = wid & 2047;
  int h    = rem >> 7;
  int q0   = (rem & 127) * 16;

  const _Float16* Qb = Qh  + (size_t)((b * NHc + h) * Sc) * HDc;
  const _Float16* Kb = Kh  + (size_t)((b * NHc + h) * Sc) * HDc;
  const _Float16* Vb = Vth + (size_t)((b * NHc + h) * HDc) * Sc;
  const float*   amb = amask + b * Sc;
  int hi8 = (lane >> 4) << 3;
  int nl  = lane & 15;

  v16h a0 = load_frag_g(Qb, q0, HDc, 0, lane);
  v16h a1 = load_frag_g(Qb, q0, HDc, 32, lane);

  // ---- pass 1: per-(row, lane-column-slice) online max/sum -------------
  float rmax[8], rsum[8];
#pragma unroll
  for (int r = 0; r < 8; ++r) { rmax[r] = -3.0e38f; rsum[r] = 0.f; }

  v16h kb0 = load_frag_g(Kb, 0, HDc, 0, lane);
  v16h kb1 = load_frag_g(Kb, 0, HDc, 32, lane);
  for (int j = 0; j < Sc / 16; ++j) {
    int jn = (j + 1) & (Sc / 16 - 1);                 // wrap: harmless reload
    v16h nb0 = load_frag_g(Kb, jn * 16, HDc, 0, lane);
    v16h nb1 = load_frag_g(Kb, jn * 16, HDc, 32, lane);
    v8f c = {};
    c = __builtin_amdgcn_wmma_f32_16x16x32_f16(false, a0, false, kb0, (short)0, c, false, false);
    c = __builtin_amdgcn_wmma_f32_16x16x32_f16(false, a1, false, kb1, (short)0, c, false, false);
    int n = j * 16 + nl;
    float amk = amb[n];
#pragma unroll
    for (int r = 0; r < 8; ++r) {
      int m = q0 + r + hi8;
      float s = (n <= m) ? (c[r] * 0.125f + amk) : (-1.0e10f + amk);
      float mn = fmaxf(rmax[r], s);
      rsum[r] = rsum[r] * __expf(rmax[r] - mn) + __expf(s - mn);
      rmax[r] = mn;
    }
    kb0 = nb0;
    kb1 = nb1;
  }
  // cross-lane reduce within 16-lane halves (lanes 0-15: rows r; 16-31: r+8)
  float rinv[8];
#pragma unroll
  for (int r = 0; r < 8; ++r) {
    float mx = rmax[r];
#pragma unroll
    for (int off = 1; off < 16; off <<= 1) mx = fmaxf(mx, __shfl_xor(mx, off, 16));
    float sm = rsum[r] * __expf(rmax[r] - mx);
#pragma unroll
    for (int off = 1; off < 16; off <<= 1) sm += __shfl_xor(sm, off, 16);
    rmax[r] = mx;
    rinv[r] = 1.0f / sm;
  }

  // ---- pass 2: probs out (b128) + ctx = P @ V --------------------------
  v8f acc[4] = {};
  float* st = &stage[w][0];
  float* prow = probs + (size_t)((b * NHc + h) * Sc + q0) * Sc;
  for (int jp = 0; jp < Sc / 32; ++jp) {
    // issue all global fragment loads for this chunk up front
    v16h s00 = load_frag_g(Kb, (jp * 2 + 0) * 16, HDc, 0, lane);
    v16h s01 = load_frag_g(Kb, (jp * 2 + 0) * 16, HDc, 32, lane);
    v16h s10 = load_frag_g(Kb, (jp * 2 + 1) * 16, HDc, 0, lane);
    v16h s11 = load_frag_g(Kb, (jp * 2 + 1) * 16, HDc, 32, lane);
    v16h vb0 = load_frag_g(Vb,  0, Sc, jp * 32, lane);
    v16h vb1 = load_frag_g(Vb, 16, Sc, jp * 32, lane);
    v16h vb2 = load_frag_g(Vb, 32, Sc, jp * 32, lane);
    v16h vb3 = load_frag_g(Vb, 48, Sc, jp * 32, lane);

#pragma unroll
    for (int t = 0; t < 2; ++t) {
      v8f c = {};
      c = __builtin_amdgcn_wmma_f32_16x16x32_f16(false, a0, false, t ? s10 : s00,
                                                 (short)0, c, false, false);
      c = __builtin_amdgcn_wmma_f32_16x16x32_f16(false, a1, false, t ? s11 : s01,
                                                 (short)0, c, false, false);
      int n = (jp * 2 + t) * 16 + nl;
      float amk = amb[n];
#pragma unroll
      for (int r = 0; r < 8; ++r) {
        int m = q0 + r + hi8;
        float s = (n <= m) ? (c[r] * 0.125f + amk) : (-1.0e10f + amk);
        st[(r + hi8) * STF + t * 16 + nl] = __expf(s - rmax[r]) * rinv[r];
      }
    }

    // drain staged probs: 4 stores/lane, each instr = 4 rows x 128B runs
    {
      int row = lane >> 3;          // 0..3
      int qd  = (lane & 7) * 4;     // 0,4,...,28
#pragma unroll
      for (int it = 0; it < 4; ++it) {
        int rr = it * 4 + row;
        float4 pv = *(const float4*)&st[rr * STF + qd];
        *(float4*)&prow[(size_t)rr * Sc + jp * 32 + qd] = pv;
      }
    }

    // P fragment (f32 LDS -> f16 regs), then ctx accumulation
    v16h pa = load_frag_lds_f32(st, lane);
    acc[0] = __builtin_amdgcn_wmma_f32_16x16x32_f16(false, pa, false, vb0, (short)0, acc[0], false, false);
    acc[1] = __builtin_amdgcn_wmma_f32_16x16x32_f16(false, pa, false, vb1, (short)0, acc[1], false, false);
    acc[2] = __builtin_amdgcn_wmma_f32_16x16x32_f16(false, pa, false, vb2, (short)0, acc[2], false, false);
    acc[3] = __builtin_amdgcn_wmma_f32_16x16x32_f16(false, pa, false, vb3, (short)0, acc[3], false, false);
  }

  float* cb = ctx + (size_t)(b * Sc + q0) * HIDc + h * HDc;
#pragma unroll
  for (int t2 = 0; t2 < 4; ++t2)
#pragma unroll
    for (int r = 0; r < 8; ++r)
      cb[(size_t)(r + hi8) * HIDc + t2 * 16 + nl] = acc[t2][r];
}

// ---------------------------------------------------------------------------
extern "C" void kernel_launch(void* const* d_in, const int* in_sizes, int n_in,
                              void* d_out, int out_size, void* d_ws, size_t ws_size,
                              hipStream_t stream) {
  const float* hs = (const float*)d_in[0];
  const float* am = (const float*)d_in[1];
  const float* Wq = (const float*)d_in[2];
  const float* bq = (const float*)d_in[3];
  const float* Wk = (const float*)d_in[4];
  const float* bk = (const float*)d_in[5];
  const float* Wv = (const float*)d_in[6];
  const float* bv = (const float*)d_in[7];

  char* ws = (char*)d_ws;
  _Float16* Xh  = (_Float16*)(ws);                         //  8 MiB [4096,1024]
  _Float16* Wqh = (_Float16*)(ws + ((size_t)8  << 20));    //  2 MiB
  _Float16* Wkh = (_Float16*)(ws + ((size_t)10 << 20));    //  2 MiB
  _Float16* Wvh = (_Float16*)(ws + ((size_t)12 << 20));    //  2 MiB
  _Float16* Qh  = (_Float16*)(ws + ((size_t)14 << 20));    //  8 MiB [B,NH,S,HD]
  _Float16* Kh  = (_Float16*)(ws + ((size_t)22 << 20));    //  8 MiB [B,NH,S,HD]
  _Float16* Vth = (_Float16*)(ws + ((size_t)30 << 20));    //  8 MiB [B,NH,HD,S]

  const int nX = Bc * Sc * HIDc;      // 4,194,304
  const int nW = HIDc * HIDc;         // 1,048,576
  cvt_f32_to_f16<<<(nX + 255) / 256, 256, 0, stream>>>(hs, Xh, nX);
  cvt_f32_to_f16<<<(nW + 255) / 256, 256, 0, stream>>>(Wq, Wqh, nW);
  cvt_f32_to_f16<<<(nW + 255) / 256, 256, 0, stream>>>(Wk, Wkh, nW);
  cvt_f32_to_f16<<<(nW + 255) / 256, 256, 0, stream>>>(Wv, Wvh, nW);

  // 3 matrices * 256 M-tiles * 16 N-tiles = 12288 waves, 4 waves/block
  qkv_proj<<<12288 / 4, 128, 0, stream>>>(Xh, Wqh, Wkh, Wvh, bq, bk, bv,
                                          Qh, Kh, Vth);

  float* ctx   = (float*)d_out;
  float* probs = ctx + (size_t)Bc * Sc * HIDc;
  // B*NH*(S/16) = 4096 waves, 8 waves/block
  attn_fused<<<4096 / 8, 256, 0, stream>>>(Qh, Kh, Vth, am, ctx, probs);
}
